// DoubleAttentionLayerA_34471407518132
// MI455X (gfx1250) — matrix operand order repaired and checked by reading, not compiled, not run on hardware
//
#include <hip/hip_runtime.h>
#include <hip/hip_bf16.h>

// ---------------- problem dims (fixed by the reference) ----------------
#define IN_CH 512
#define C_M   512
#define C_N   128
#define BATCH 16
#define HW    3136   // 56*56, K=1 so N_spatial == HW

typedef __attribute__((ext_vector_type(16))) __bf16 v16bf;
typedef __attribute__((ext_vector_type(8)))  __bf16 v8bf;
typedef __attribute__((ext_vector_type(8)))  float  v8f;

// ---------------------------------------------------------------------
// f32 -> bf16 elementwise convert (weights)
// ---------------------------------------------------------------------
__global__ void k_convert(const float* __restrict__ s, __bf16* __restrict__ d, int n) {
    int i = blockIdx.x * blockDim.x + threadIdx.x;
    if (i < n) d[i] = (__bf16)s[i];
}

// ---------------------------------------------------------------------
// x[b][c][p] (f32) -> xT[b][p][c] (bf16): LDS-tiled transpose+convert.
// 32x32 tiles, block = (32,8). HW and IN_CH divide by 32 exactly.
// ---------------------------------------------------------------------
__global__ __launch_bounds__(256)
void k_transpose_convert(const float* __restrict__ x, __bf16* __restrict__ xT) {
    __shared__ __bf16 tile[32][33];     // pad to break bank conflicts
    const int bz = blockIdx.z;
    const int p0 = blockIdx.x * 32;     // pixel tile
    const int c0 = blockIdx.y * 32;     // channel tile
    const float* xb = x + (size_t)bz * IN_CH * HW;
    __bf16* xTb = xT + (size_t)bz * IN_CH * HW;
    const int tx = threadIdx.x;         // 0..31
    const int ty = threadIdx.y;         // 0..7

    #pragma unroll
    for (int i = 0; i < 4; ++i) {
        int c = ty + 8 * i;
        tile[c][tx] = (__bf16)xb[(size_t)(c0 + c) * HW + p0 + tx];  // coalesced read
    }
    __syncthreads();
    #pragma unroll
    for (int i = 0; i < 4; ++i) {
        int p = ty + 8 * i;
        xTb[(size_t)(p0 + p) * IN_CH + c0 + tx] = tile[tx][p];      // coalesced write
    }
}

// ---------------------------------------------------------------------
// Unified NT GEMM:  Y[b] = A[b] (MxK, row-major) * Bt[b]^T (Bt is NxK row-major)
// Per wave: (MI*16) x (NJ*16) output tile; MI x NJ WMMAs per K-step of 32.
// grid = (N/(NJ*16), M/(MI*16), BATCH), block = 32 (one wave; EXEC all ones)
// ---------------------------------------------------------------------
template <int MI, int NJ>
__global__ __launch_bounds__(32)
void k_gemm_nt(const __bf16* __restrict__ A,  long long sAb,
               const __bf16* __restrict__ Bt, long long sBb,
               const float*  __restrict__ bias,
               float* __restrict__ Yf, __bf16* __restrict__ Yb, long long sYb,
               int M, int N, int K)
{
    const int lane = threadIdx.x;
    const int half = lane >> 4;   // 0 or 1
    const int l16  = lane & 15;
    const int bz = blockIdx.z;
    const int m0 = blockIdx.y * (MI * 16);
    const int n0 = blockIdx.x * (NJ * 16);

    const __bf16* Ab  = A  + (size_t)bz * (size_t)sAb;
    const __bf16* Btb = Bt + (size_t)bz * (size_t)sBb;

    // A fragment (16x32 bf16): lane holds row; elems 0..7 -> K = half*8 + 0..7,
    // elems 8..15 -> K = half*8 + 16..23  (ISA 16-bit A layout)
    const __bf16* aRow[MI];
    #pragma unroll
    for (int i = 0; i < MI; ++i)
        aRow[i] = Ab + (size_t)(m0 + 16 * i + l16) * K + half * 8;
    // B fragment (32x16 bf16): lane holds column; same K layout as A (half*8 + 0..7, half*8 + 16..23)
    const __bf16* bRow[NJ];
    #pragma unroll
    for (int j = 0; j < NJ; ++j)
        bRow[j] = Btb + (size_t)(n0 + 16 * j + l16) * K + half * 8;

    v8f acc[MI][NJ];
    #pragma unroll
    for (int i = 0; i < MI; ++i)
        #pragma unroll
        for (int j = 0; j < NJ; ++j)
            acc[i][j] = (v8f){};

    union F { v16bf v; v8bf h[2]; };

    for (int k0 = 0; k0 < K; k0 += 32) {
        F fa[MI], fb[NJ];
        #pragma unroll
        for (int i = 0; i < MI; ++i) {
            fa[i].h[0] = *(const v8bf*)(aRow[i] + k0);
            fa[i].h[1] = *(const v8bf*)(aRow[i] + k0 + 16);
        }
        #pragma unroll
        for (int j = 0; j < NJ; ++j) {
            fb[j].h[0] = *(const v8bf*)(bRow[j] + k0);
            fb[j].h[1] = *(const v8bf*)(bRow[j] + k0 + 16);
            // stream-ahead prefetch (speculative; safe past end): 8 K-steps = 512B
            __builtin_prefetch(bRow[j] + k0 + 256, 0, 1);
        }
        #pragma unroll
        for (int i = 0; i < MI; ++i)
            #pragma unroll
            for (int j = 0; j < NJ; ++j)
                acc[i][j] = __builtin_amdgcn_wmma_f32_16x16x32_bf16(
                    false, fa[i].v, false, fb[j].v, (short)0, acc[i][j], false, false);
    }

    // C/D layout: lane 0-15: VGPR r -> M=r, N=lane ; lane 16-31: VGPR r -> M=8+r, N=lane-16
    float*  yfb = Yf ? (Yf + (size_t)bz * (size_t)sYb) : nullptr;
    __bf16* ybb = Yb ? (Yb + (size_t)bz * (size_t)sYb) : nullptr;
    #pragma unroll
    for (int i = 0; i < MI; ++i) {
        #pragma unroll
        for (int r = 0; r < 8; ++r) {
            int m = m0 + 16 * i + r + half * 8;
            float add = bias ? bias[m] : 0.0f;
            #pragma unroll
            for (int j = 0; j < NJ; ++j) {
                float v = acc[i][j][r] + add;
                size_t o = (size_t)m * N + n0 + 16 * j + l16;
                if (yfb) yfb[o] = v;
                if (ybb) ybb[o] = (__bf16)v;
            }
        }
    }
}

// ---------------------------------------------------------------------
// Softmax over a row of `len` f32, bf16 out. One block (256 thr) per row.
// Rows = BATCH*C_N, len = HW. LDS tree reductions.
// ---------------------------------------------------------------------
__global__ __launch_bounds__(256)
void k_softmax_row(const float* __restrict__ X, __bf16* __restrict__ Y, int len)
{
    __shared__ float red[256];
    const size_t row = blockIdx.x;
    const float* xr = X + row * (size_t)len;
    __bf16* yr = Y + row * (size_t)len;
    int t = threadIdx.x;

    float mx = -3.4e38f;
    for (int i = t; i < len; i += 256) mx = fmaxf(mx, xr[i]);
    red[t] = mx; __syncthreads();
    for (int s = 128; s > 0; s >>= 1) {
        if (t < s) red[t] = fmaxf(red[t], red[t + s]);
        __syncthreads();
    }
    mx = red[0]; __syncthreads();

    float sum = 0.0f;
    for (int i = t; i < len; i += 256) sum += __expf(xr[i] - mx);
    red[t] = sum; __syncthreads();
    for (int s = 128; s > 0; s >>= 1) {
        if (t < s) red[t] += red[t + s];
        __syncthreads();
    }
    float inv = 1.0f / red[0];
    for (int i = t; i < len; i += 256) yr[i] = (__bf16)(__expf(xr[i] - mx) * inv);
}

// ---------------------------------------------------------------------
// Per-pixel softmax over C_N channels of V[b][n][p]; writes TRANSPOSED
// rows Yt[b][p][n] (bf16) so the Z-GEMM B-operand is K-contiguous.
// ---------------------------------------------------------------------
__global__ __launch_bounds__(256)
void k_softmax_col_t(const float* __restrict__ V, __bf16* __restrict__ Yt)
{
    int idx = blockIdx.x * blockDim.x + threadIdx.x;   // over BATCH*HW pixels
    if (idx >= BATCH * HW) return;
    int bz = idx / HW, p = idx - bz * HW;
    const float* vp = V + (size_t)bz * C_N * HW + p;
    __bf16* yp = Yt + (size_t)bz * HW * C_N + (size_t)p * C_N;

    float mx = -3.4e38f;
    #pragma unroll 4
    for (int n = 0; n < C_N; ++n) mx = fmaxf(mx, vp[(size_t)n * HW]);
    float s = 0.0f;
    #pragma unroll 4
    for (int n = 0; n < C_N; ++n) s += __expf(vp[(size_t)n * HW] - mx);
    float inv = 1.0f / s;
    #pragma unroll 4
    for (int n = 0; n < C_N; ++n) yp[n] = (__bf16)(__expf(vp[(size_t)n * HW] - mx) * inv);
}

// ---------------------------------------------------------------------
// Host-side launch
// ---------------------------------------------------------------------
extern "C" void kernel_launch(void* const* d_in, const int* in_sizes, int n_in,
                              void* d_out, int out_size, void* d_ws, size_t ws_size,
                              hipStream_t stream)
{
    const float* x  = (const float*)d_in[0];   // (16,512,56,56)
    const float* wA = (const float*)d_in[1];   // (512,512)
    const float* bA = (const float*)d_in[2];   // (512,)
    const float* wB = (const float*)d_in[3];   // (128,512)
    const float* bB = (const float*)d_in[4];   // (128,)
    const float* wV = (const float*)d_in[5];   // (128,512)
    const float* bV = (const float*)d_in[6];   // (128,)
    float* out = (float*)d_out;                // (16,512,56,56) f32

    // ---- workspace layout (bytes, 256-aligned blocks) ----
    char* ws = (char*)d_ws;
    size_t off = 0;
    auto take = [&](size_t bytes) { char* p = ws + off; off = (off + bytes + 255) & ~(size_t)255; return p; };

    const size_t xT_e   = (size_t)BATCH * IN_CH * HW;        // bf16
    const size_t A_e    = (size_t)BATCH * C_M * HW;          // bf16
    const size_t BV_e   = (size_t)BATCH * C_N * HW;          // f32 each
    const size_t att_e  = (size_t)BATCH * C_N * HW;          // bf16 each
    const size_t G_e    = (size_t)BATCH * C_M * C_N;         // bf16

    __bf16* xT   = (__bf16*)take(xT_e * 2);
    __bf16* tmpA = (__bf16*)take(A_e * 2);
    float*  Bf   = (float*) take(BV_e * 4);
    float*  Vf   = (float*) take(BV_e * 4);
    __bf16* sB   = (__bf16*)take(att_e * 2);
    __bf16* sVt  = (__bf16*)take(att_e * 2);
    __bf16* Gb   = (__bf16*)take(G_e * 2);
    __bf16* wAb  = (__bf16*)take((size_t)C_M * IN_CH * 2);
    __bf16* wBb  = (__bf16*)take((size_t)C_N * IN_CH * 2);
    __bf16* wVb  = (__bf16*)take((size_t)C_N * IN_CH * 2);
    (void)ws_size; (void)in_sizes; (void)n_in; (void)out_size;

    // 1) convert weights to bf16
    k_convert<<<dim3((C_M * IN_CH + 255) / 256), 256, 0, stream>>>(wA, wAb, C_M * IN_CH);
    k_convert<<<dim3((C_N * IN_CH + 255) / 256), 256, 0, stream>>>(wB, wBb, C_N * IN_CH);
    k_convert<<<dim3((C_N * IN_CH + 255) / 256), 256, 0, stream>>>(wV, wVb, C_N * IN_CH);

    // 2) transpose+convert x -> xT[b][p][c] bf16 (LDS-tiled, coalesced both sides)
    k_transpose_convert<<<dim3(HW / 32, IN_CH / 32, BATCH), dim3(32, 8), 0, stream>>>(x, xT);

    const long long sX = (long long)IN_CH * HW;   // xT per-batch stride (elements)

    // 3) conv A: tmpA[b][m][p] (bf16) = wA * x + bA      [64x64 wave tiles]
    k_gemm_nt<4, 4><<<dim3(HW / 64, C_M / 64, BATCH), 32, 0, stream>>>(
        wAb, 0, xT, sX, bA, nullptr, tmpA, (long long)C_M * HW, C_M, HW, IN_CH);

    // 4) conv B: Bf[b][n][p] (f32)
    k_gemm_nt<4, 4><<<dim3(HW / 64, C_N / 64, BATCH), 32, 0, stream>>>(
        wBb, 0, xT, sX, bB, Bf, nullptr, (long long)C_N * HW, C_N, HW, IN_CH);

    // 5) conv V: Vf[b][n][p] (f32)
    k_gemm_nt<4, 4><<<dim3(HW / 64, C_N / 64, BATCH), 32, 0, stream>>>(
        wVb, 0, xT, sX, bV, Vf, nullptr, (long long)C_N * HW, C_N, HW, IN_CH);

    // 6) softmax over pixels for B -> sB[b][n][p] bf16
    k_softmax_row<<<dim3(BATCH * C_N), 256, 0, stream>>>(Bf, sB, HW);

    // 7) per-pixel softmax over channels for V -> sVt[b][p][n] bf16
    k_softmax_col_t<<<dim3((BATCH * HW + 255) / 256), 256, 0, stream>>>(Vf, sVt);

    // 8) G[b][m][n] = sum_p tmpA[m][p] * sB[n][p]   (NT, K=HW)
    //    Small output (512x128): use 32x32 tiles for 4x more waves (machine fill).
    k_gemm_nt<2, 2><<<dim3(C_N / 32, C_M / 32, BATCH), 32, 0, stream>>>(
        tmpA, (long long)C_M * HW, sB, (long long)C_N * HW, nullptr,
        nullptr, Gb, (long long)C_M * C_N, C_M, C_N, HW);

    // 9) Z[b][m][p] = sum_n G[m][n] * sVt[p][n]     (NT, K=C_N) -> d_out f32
    k_gemm_nt<4, 4><<<dim3(HW / 64, C_M / 64, BATCH), 32, 0, stream>>>(
        Gb, (long long)C_M * C_N, sVt, (long long)HW * C_N, nullptr,
        out, nullptr, (long long)C_M * HW, C_M, HW, C_N);
}
